// KANLinear_17051020165311
// MI455X (gfx1250) — matrix operand order repaired and checked by reading, not compiled, not run on hardware
//
#include <hip/hip_runtime.h>
#include <hip/hip_bf16.h>

// ---------------------------------------------------------------------------
// KAN linear:  out = silu(x) @ Wb^T + einsum('oik,bik->bo', Ws, bspline(x))
// One bf16 GEMM with K = 9*256 = 2304:
//   kk = k*256 + i :  k in [0,8) -> basis_k(x[b,i]),  k==8 -> silu(x[b,i])
//   Wt[o,kk]       :  k in [0,8) -> Ws[o,i,k],        k==8 -> Wb[o,i]
// GEMM: v_wmma_f32_16x16x32_bf16, double-buffered LDS fed by
// global_load_async_to_lds_b128 (ASYNCcnt) instead of load+ds_store.
// ---------------------------------------------------------------------------

typedef __attribute__((ext_vector_type(16))) __bf16 v16bf;
typedef __attribute__((ext_vector_type(8)))  __bf16 v8bf;
typedef __attribute__((ext_vector_type(8)))  float  v8f;

#define IN_F   256
#define OUT_F  256
#define KTOT   2304   // 9 * 256 = 72 * 32
#define BM     128
#define BN     128
#define BK     32
#define NKB    (KTOT / BK)   // 72
#define LDT    40     // padded LDS row stride (elements) to break bank conflicts

static __device__ __forceinline__ unsigned short f2bf(float f) {
    union { float f; unsigned u; } v; v.f = f;
    unsigned r = v.u + 0x7FFFu + ((v.u >> 16) & 1u);   // round-to-nearest-even
    return (unsigned short)(r >> 16);
}

// async 16B global -> LDS copy (no VGPR round trip, tracked by ASYNCcnt)
static __device__ __forceinline__ void async_cp16(unsigned lds_off,
                                                  const unsigned short* gptr) {
    asm volatile("global_load_async_to_lds_b128 %0, %1, off"
                 :
                 : "v"(lds_off), "v"((unsigned long long)(uintptr_t)gptr)
                 : "memory");
}

// --- pack spline_weight (o,i,8) + base_weight (o,i) -> bf16 Wt[o, k*256+i] --
__global__ void kan_wpack(const float* __restrict__ sw,
                          const float* __restrict__ bw,
                          unsigned short* __restrict__ Wt) {
    int o = blockIdx.x;            // 256
    int i = threadIdx.x;           // 256
    const float* swp = sw + ((size_t)o * IN_F + i) * 8;
    size_t base = (size_t)o * KTOT + i;
#pragma unroll
    for (int k = 0; k < 8; ++k)
        Wt[base + (size_t)k * IN_F] = f2bf(swp[k]);
    Wt[base + (size_t)8 * IN_F] = f2bf(bw[(size_t)o * IN_F + i]);
}

// --- per-element cubic B-spline basis + silu -> bf16 A[b, k*256+i] ---------
__global__ void kan_prep(const float* __restrict__ x,
                         const float* __restrict__ grid,
                         unsigned short* __restrict__ A) {
    int b = blockIdx.x;
    int i = threadIdx.x;           // 256, coalesced stores at stride 256
    float xv = x[(size_t)b * IN_F + i];

    float t[12];
#pragma unroll
    for (int j = 0; j < 12; ++j) t[j] = grid[i * 12 + j];

    float bs[11];
#pragma unroll
    for (int j = 0; j < 11; ++j)
        bs[j] = (xv >= t[j] && xv < t[j + 1]) ? 1.0f : 0.0f;

    const float EPS = 1e-8f;
#pragma unroll
    for (int p = 1; p <= 3; ++p) {
#pragma unroll
        for (int j = 0; j < 10; ++j) {
            if (j < 11 - p) {
                float left  = (xv - t[j])         / (t[j + p] - t[j] + EPS)         * bs[j];
                float right = (t[j + p + 1] - xv) / (t[j + p + 1] - t[j + 1] + EPS) * bs[j + 1];
                bs[j] = left + right;
            }
        }
    }
    float silu = xv / (1.0f + __expf(-xv));

    size_t base = (size_t)b * KTOT + i;
#pragma unroll
    for (int k = 0; k < 8; ++k)
        A[base + (size_t)k * IN_F] = f2bf(bs[k]);
    A[base + (size_t)8 * IN_F] = f2bf(silu);
}

// --- bf16 WMMA GEMM: out[B,256] = A[B,2304] * Wt[256,2304]^T ---------------
union FragAB { v16bf v; v8bf h[2]; };

__global__ __launch_bounds__(256) void kan_gemm(const unsigned short* __restrict__ A,
                                                const unsigned short* __restrict__ Wt,
                                                float* __restrict__ out) {
    __shared__ unsigned short lA[2][BM * LDT];
    __shared__ unsigned short lB[2][BN * LDT];

    const int tid  = threadIdx.x;
    const int lane = tid & 31;
    const int wave = tid >> 5;         // 0..7
    const int half = lane >> 4;        // 0/1
    const int l15  = lane & 15;
    const int wm   = wave & 1;         // 2 M strips of 64
    const int wn   = wave >> 1;        // 4 N strips of 32
    const int bm   = blockIdx.x;
    const int bn   = blockIdx.y;

    // staging ownership: thread t moves rows (t>>2) and (t>>2)+64, 16B seg t&3
    const int r0  = tid >> 2;
    const int seg = tid & 3;
    const unsigned short* gA0 = A  + (size_t)(bm * BM + r0)      * KTOT + seg * 8;
    const unsigned short* gA1 = A  + (size_t)(bm * BM + r0 + 64) * KTOT + seg * 8;
    const unsigned short* gB0 = Wt + (size_t)(bn * BN + r0)      * KTOT + seg * 8;
    const unsigned short* gB1 = Wt + (size_t)(bn * BN + r0 + 64) * KTOT + seg * 8;

    const unsigned ldsA0[2] = {
        (unsigned)(uintptr_t)&lA[0][r0 * LDT + seg * 8],
        (unsigned)(uintptr_t)&lA[1][r0 * LDT + seg * 8] };
    const unsigned ldsA1[2] = {
        (unsigned)(uintptr_t)&lA[0][(r0 + 64) * LDT + seg * 8],
        (unsigned)(uintptr_t)&lA[1][(r0 + 64) * LDT + seg * 8] };
    const unsigned ldsB0[2] = {
        (unsigned)(uintptr_t)&lB[0][r0 * LDT + seg * 8],
        (unsigned)(uintptr_t)&lB[1][r0 * LDT + seg * 8] };
    const unsigned ldsB1[2] = {
        (unsigned)(uintptr_t)&lB[0][(r0 + 64) * LDT + seg * 8],
        (unsigned)(uintptr_t)&lB[1][(r0 + 64) * LDT + seg * 8] };

    v8f acc[4][2] = {};                // 4 M-tiles x 2 N-tiles of 16x16

    // prologue: slab 0 -> buffer 0
    async_cp16(ldsA0[0], gA0);
    async_cp16(ldsA1[0], gA1);
    async_cp16(ldsB0[0], gB0);
    async_cp16(ldsB1[0], gB1);

    for (int it = 0; it < NKB; ++it) {
        const int cur = it & 1;
        if (it + 1 < NKB) {            // prefetch next slab into the other buffer
            const int nxt = cur ^ 1;
            const int ko  = (it + 1) * BK;
            async_cp16(ldsA0[nxt], gA0 + ko);
            async_cp16(ldsA1[nxt], gA1 + ko);
            async_cp16(ldsB0[nxt], gB0 + ko);
            async_cp16(ldsB1[nxt], gB1 + ko);
            asm volatile("s_wait_asynccnt 0x4" ::: "memory");  // slab `it` landed
        } else {
            asm volatile("s_wait_asynccnt 0x0" ::: "memory");
        }
        __syncthreads();               // all waves' slab data visible in LDS

        // A fragment (16x32 bf16): lane row = M; elems 0..7 -> K = 8*half + e,
        //                                      elems 8..15 -> K = 16 + 8*half + e
        FragAB fa[4];
#pragma unroll
        for (int mt = 0; mt < 4; ++mt) {
            const unsigned short* rp = &lA[cur][(wm * 64 + mt * 16 + l15) * LDT];
            fa[mt].h[0] = *(const v8bf*)(rp + 8 * half);
            fa[mt].h[1] = *(const v8bf*)(rp + 16 + 8 * half);
        }
        // B fragment (32x16 bf16), N-major in LDS: lane col = N,
        //   elems 0..7 -> K = 8*half + e, elems 8..15 -> K = 16 + 8*half + e
        FragAB fb[2];
#pragma unroll
        for (int nt = 0; nt < 2; ++nt) {
            const unsigned short* rp = &lB[cur][(wn * 32 + nt * 16 + l15) * LDT];
            fb[nt].h[0] = *(const v8bf*)(rp + 8 * half);
            fb[nt].h[1] = *(const v8bf*)(rp + 16 + 8 * half);
        }

#pragma unroll
        for (int mt = 0; mt < 4; ++mt)
#pragma unroll
            for (int nt = 0; nt < 2; ++nt)
                acc[mt][nt] = __builtin_amdgcn_wmma_f32_16x16x32_bf16(
                    false, fa[mt].v, false, fb[nt].v,
                    (short)0, acc[mt][nt], false, false);

        __syncthreads();               // reads done before this buffer is refilled
    }

    // C/D layout: VGPR r -> row M = r + 8*half, col N = l15
#pragma unroll
    for (int mt = 0; mt < 4; ++mt) {
#pragma unroll
        for (int nt = 0; nt < 2; ++nt) {
            int ocol  = bn * BN + wn * 32 + nt * 16 + l15;
            int mrow0 = bm * BM + wm * 64 + mt * 16 + half * 8;
#pragma unroll
            for (int r = 0; r < 8; ++r)
                out[(size_t)(mrow0 + r) * OUT_F + ocol] = acc[mt][nt][r];
        }
    }
}

extern "C" void kernel_launch(void* const* d_in, const int* in_sizes, int n_in,
                              void* d_out, int out_size, void* d_ws, size_t ws_size,
                              hipStream_t stream) {
    const float* x    = (const float*)d_in[0];   // (batch, 256)
    const float* grid = (const float*)d_in[1];   // (256, 12)
    const float* sw   = (const float*)d_in[2];   // (256, 256, 8)
    const float* bw   = (const float*)d_in[3];   // (256, 256)
    float* out        = (float*)d_out;

    const int batch = in_sizes[0] / IN_F;        // 32768

    unsigned short* Aws = (unsigned short*)d_ws;              // batch x 2304 bf16
    unsigned short* Wtw = Aws + (size_t)batch * KTOT;         // 256 x 2304 bf16

    kan_wpack<<<OUT_F, IN_F, 0, stream>>>(sw, bw, Wtw);
    kan_prep <<<batch, IN_F, 0, stream>>>(x, grid, Aws);

    dim3 g(batch / BM, OUT_F / BN);
    kan_gemm<<<g, 256, 0, stream>>>(Aws, Wtw, out);
}
